// UnifiedTransformerHead_4621384810898
// MI455X (gfx1250) — compile-verified, matrix operand order checked by reading, not run on hardware
//
#include <hip/hip_runtime.h>
#include <hip/hip_bf16.h>

typedef _Float16 f16;
typedef __bf16   bf16;
typedef __attribute__((ext_vector_type(16))) _Float16 v16h;
typedef __attribute__((ext_vector_type(16))) __bf16   v16bf;
typedef __attribute__((ext_vector_type(8)))  float    v8f;
typedef __attribute__((ext_vector_type(4)))  unsigned int u32x4;
typedef __attribute__((ext_vector_type(8)))  int      i32x8;
typedef __attribute__((ext_vector_type(4)))  int      i32x4;

#define DMODEL 256
#define NHEADS 8
#define DH     32
#define NQT    515     // NSEG + NDET + NPT
#define LVTOT  1045    // 784 + 196 + 49 + 16

// ---------------------------------------------------------------------------
// Tensor Data Mover: 2-D tile load Global -> LDS (f32 elements).
// D# packing per CDNA5 ISA.
// pad_interval=4 (pad every 32 DWORDs), pad_amount=3 (4 DWORDs) => LDS row
// stride = 36 floats (bank-conflict-free fragment reads).
// Rows beyond tens_d1 are hardware zero-filled (free edge handling).
// ---------------------------------------------------------------------------
__device__ __forceinline__ unsigned lds_off(const void* p)
{
    // generic pointer to LDS: low 32 bits are the LDS byte offset
    return (unsigned)(unsigned long long)(uintptr_t)p;
}

__device__ __forceinline__ void tdm_load_2d_f32(unsigned lds_addr, const void* gaddr,
                                                unsigned tile_d0, unsigned tile_d1,
                                                unsigned tens_d0, unsigned tens_d1,
                                                unsigned stride_elems)
{
    unsigned long long ga = (unsigned long long)(uintptr_t)gaddr;
    u32x4 g0;
    g0[0] = 1u;                                              // count=1 (valid), user mode
    g0[1] = lds_addr;                                        // lds_addr [63:32]
    g0[2] = (unsigned)(ga & 0xffffffffu);                    // global_addr[31:0]
    g0[3] = (unsigned)((ga >> 32) & 0x01ffffffu) | (2u << 30); // addr[56:32] | type=2
    i32x8 g1;
    g1[0] = (int)((2u << 16)        // data_size = 4 bytes
                | (1u << 20)        // pad_enable
                | (4u << 22)        // pad_interval: every 32 DWORDs
                | (3u << 25));      // pad_amount: 4 DWORDs
    g1[1] = (int)((tens_d0 & 0xffffu) << 16);                               // dim0[15:0]
    g1[2] = (int)(((tens_d0 >> 16) & 0xffffu) | ((tens_d1 & 0xffffu) << 16)); // dim0[31:16]|dim1[15:0]
    g1[3] = (int)(((tens_d1 >> 16) & 0xffffu) | ((tile_d0 & 0xffffu) << 16)); // dim1[31:16]|tile_dim0
    g1[4] = (int)(tile_d1 & 0xffffu);                        // tile_dim1 (tile_dim2 = 0)
    g1[5] = (int)stride_elems;                               // tensor_dim0_stride[31:0]
    g1[6] = 0;                                               // stride[47:32] | dim1_stride[15:0]
    g1[7] = 0;
    i32x4 z4 = {0, 0, 0, 0};
    i32x8 z8 = {0, 0, 0, 0, 0, 0, 0, 0};
    __builtin_amdgcn_tensor_load_to_lds(g0, g1, z4, z4, z8, 0);
}

// ---------------------------------------------------------------------------
// GEMM: Y[M,N] = X[M,K] @ W[N,K]^T + bias, optional ReLU.  bf16 WMMA, f32 acc.
// 64x64 tile per block; 128 threads = 4 waves, each wave owns a 32x32 output
// tile (4 WMMAs per K-step on 4 fragment gathers -> 1.0 gathers/WMMA).
// TDM double-buffered tile feed; fragments converted f32->bf16 on LDS read.
// ---------------------------------------------------------------------------
__launch_bounds__(128)
__global__ void linear_wmma_bf16(const float* __restrict__ X,
                                 const float* __restrict__ W,
                                 const float* __restrict__ bias,
                                 float* __restrict__ Y,
                                 int M, int N, int K, int relu)
{
    __shared__ float Asf[2][64][36];
    __shared__ float Wsf[2][64][36];

    const int m0   = blockIdx.x * 64;
    const int n0   = blockIdx.y * 64;
    const int tid  = threadIdx.x;
    const int lane = tid & 31;
    const int wid  = tid >> 5;         // 0..3
    const int wm   = (wid & 1) * 32;   // 32-row strip within the 64-row tile
    const int wn   = (wid >> 1) * 32;  // 32-col strip within the 64-col tile

    const int mrow  = lane & 15;
    const int khalf = (lane >> 4) * 8; // ISA 16-bit A/B fragment K-half select

    v8f c00 = {}, c01 = {}, c10 = {}, c11 = {};

    // prime buffer 0
    if (wid == 0) {
        tdm_load_2d_f32(lds_off(&Asf[0][0][0]), X + (size_t)m0 * K,
                        32, 64, (unsigned)K, (unsigned)(M - m0), (unsigned)K);
        tdm_load_2d_f32(lds_off(&Wsf[0][0][0]), W + (size_t)n0 * K,
                        32, 64, (unsigned)K, (unsigned)(N - n0), (unsigned)K);
    }

    int buf = 0;
    for (int k0 = 0; k0 < K; k0 += 32) {
        if (wid == 0) {
            if (k0 + 32 < K) {  // issue next tile, then wait for current (<=2 in flight)
                tdm_load_2d_f32(lds_off(&Asf[buf ^ 1][0][0]),
                                X + (size_t)m0 * K + (k0 + 32),
                                32, 64, (unsigned)(K - k0 - 32), (unsigned)(M - m0),
                                (unsigned)K);
                tdm_load_2d_f32(lds_off(&Wsf[buf ^ 1][0][0]),
                                W + (size_t)n0 * K + (k0 + 32),
                                32, 64, (unsigned)(K - k0 - 32), (unsigned)(N - n0),
                                (unsigned)K);
                __builtin_amdgcn_s_wait_tensorcnt(2);
            } else {
                __builtin_amdgcn_s_wait_tensorcnt(0);
            }
        }
        __syncthreads();

        // gather fragments per CDNA5 16-bit WMMA VGPR layout (ISA 7.12.2),
        // converting f32 -> bf16 on read (v_cvt_pk_bf16_f32 co-issues with WMMA)
        v16bf a0, a1, b0, b1;
#pragma unroll
        for (int e = 0; e < 16; ++e) {
            int kk = ((e >> 3) * 16) + khalf + (e & 7);
            a0[e] = (bf16)Asf[buf][wm + mrow][kk];
            a1[e] = (bf16)Asf[buf][wm + 16 + mrow][kk];
            b0[e] = (bf16)Wsf[buf][wn + mrow][kk];
            b1[e] = (bf16)Wsf[buf][wn + 16 + mrow][kk];
        }
        c00 = __builtin_amdgcn_wmma_f32_16x16x32_bf16(false, a0, false, b0,
                                                      (short)0, c00, false, false);
        c01 = __builtin_amdgcn_wmma_f32_16x16x32_bf16(false, a0, false, b1,
                                                      (short)0, c01, false, false);
        c10 = __builtin_amdgcn_wmma_f32_16x16x32_bf16(false, a1, false, b0,
                                                      (short)0, c10, false, false);
        c11 = __builtin_amdgcn_wmma_f32_16x16x32_bf16(false, a1, false, b1,
                                                      (short)0, c11, false, false);
        __syncthreads();
        buf ^= 1;
    }

    // epilogue: C/D layout -> lane holds col n=lane&15, rows r + 8*(lane>=16)
    const int nloc = lane & 15;
    const int mofs = (lane >> 4) * 8;
#pragma unroll
    for (int r = 0; r < 8; ++r) {
        int gm0 = m0 + wm + r + mofs;
        int gm1 = gm0 + 16;
        int gn0 = n0 + wn + nloc;
        int gn1 = gn0 + 16;
        if (gm0 < M) {
            if (gn0 < N) {
                float v = c00[r] + bias[gn0];
                if (relu) v = v > 0.0f ? v : 0.0f;
                Y[(size_t)gm0 * N + gn0] = v;
            }
            if (gn1 < N) {
                float v = c01[r] + bias[gn1];
                if (relu) v = v > 0.0f ? v : 0.0f;
                Y[(size_t)gm0 * N + gn1] = v;
            }
        }
        if (gm1 < M) {
            if (gn0 < N) {
                float v = c10[r] + bias[gn0];
                if (relu) v = v > 0.0f ? v : 0.0f;
                Y[(size_t)gm1 * N + gn0] = v;
            }
            if (gn1 < N) {
                float v = c11[r] + bias[gn1];
                if (relu) v = v > 0.0f ? v : 0.0f;
                Y[(size_t)gm1 * N + gn1] = v;
            }
        }
    }
}

// ---------------------------------------------------------------------------
// Flash attention, head-major [H][L][32] layout, Dh=32 == WMMA K-dim.
// One wave handles 16 queries of one head; loops 32 keys per iteration:
//   TDM K/V tile fetch, 2x wmma (S = Q K^T), online softmax, 2x wmma (O += P V).
// ---------------------------------------------------------------------------
__launch_bounds__(32)
__global__ void flash_attn_wmma(const float* __restrict__ Q,
                                const float* __restrict__ Kk,
                                const float* __restrict__ V,
                                float* __restrict__ O,
                                int Lq, int Lk, float scale)
{
    __shared__ float Ktf[32][36];
    __shared__ float Vtf[32][36];
    __shared__ f16   Pt[16][36];

    const int h  = blockIdx.y;
    const int q0 = blockIdx.x * 16;
    const float* Qh = Q  + (size_t)h * Lq * DH;
    const float* Kh = Kk + (size_t)h * Lk * DH;
    const float* Vh = V  + (size_t)h * Lk * DH;
    float*       Oh = O  + (size_t)h * Lq * DH;

    const int lane  = threadIdx.x;
    const int mrow  = lane & 15;
    const int khalf = (lane >> 4) * 8;

    // Q fragment (16x32), scale folded in
    v16h qa;
    {
        int gq = q0 + mrow;
        if (gq >= Lq) gq = Lq - 1;
#pragma unroll
        for (int e = 0; e < 16; ++e) {
            int kk = ((e >> 3) * 16) + khalf + (e & 7);
            qa[e] = (f16)(Qh[(size_t)gq * DH + kk] * scale);
        }
    }

    float m_i[8], l_i[8];
    v8f o0 = {};
    v8f o1 = {};
#pragma unroll
    for (int r = 0; r < 8; ++r) { m_i[r] = -1e30f; l_i[r] = 0.0f; }

    for (int kb = 0; kb < Lk; kb += 32) {
        // TDM: fetch 32 keys of K and V (zero-filled past Lk)
        tdm_load_2d_f32(lds_off(&Ktf[0][0]), Kh + (size_t)kb * DH,
                        32, 32, DH, (unsigned)(Lk - kb), DH);
        tdm_load_2d_f32(lds_off(&Vtf[0][0]), Vh + (size_t)kb * DH,
                        32, 32, DH, (unsigned)(Lk - kb), DH);
        __builtin_amdgcn_s_wait_tensorcnt(0);
        __syncthreads();

        // S = Q K^T for 32 keys: B[k][n] = K[key n][chan k]
        v16h b0, b1;
#pragma unroll
        for (int e = 0; e < 16; ++e) {
            int kk = ((e >> 3) * 16) + khalf + (e & 7);
            b0[e] = (f16)Ktf[mrow][kk];
            b1[e] = (f16)Ktf[16 + mrow][kk];
        }
        v8f s0 = {};
        v8f s1 = {};
        s0 = __builtin_amdgcn_wmma_f32_16x16x32_f16(false, qa, false, b0,
                                                    (short)0, s0, false, false);
        s1 = __builtin_amdgcn_wmma_f32_16x16x32_f16(false, qa, false, b1,
                                                    (short)0, s1, false, false);

        // tail mask (key index = kb + n, n = lane&15 / +16)
        int key0 = kb + (lane & 15);
        if (key0 >= Lk) {
#pragma unroll
            for (int r = 0; r < 8; ++r) s0[r] = -1e30f;
        }
        if (key0 + 16 >= Lk) {
#pragma unroll
            for (int r = 0; r < 8; ++r) s1[r] = -1e30f;
        }

        // online softmax per row; reductions stay inside each 16-lane half
#pragma unroll
        for (int r = 0; r < 8; ++r) {
            float mx = fmaxf(s0[r], s1[r]);
            mx = fmaxf(mx, __shfl_xor(mx, 8, 32));
            mx = fmaxf(mx, __shfl_xor(mx, 4, 32));
            mx = fmaxf(mx, __shfl_xor(mx, 2, 32));
            mx = fmaxf(mx, __shfl_xor(mx, 1, 32));
            float mnew = fmaxf(m_i[r], mx);
            float corr = __expf(m_i[r] - mnew);
            float p0 = __expf(s0[r] - mnew);
            float p1 = __expf(s1[r] - mnew);
            float ps = p0 + p1;
            ps += __shfl_xor(ps, 8, 32);
            ps += __shfl_xor(ps, 4, 32);
            ps += __shfl_xor(ps, 2, 32);
            ps += __shfl_xor(ps, 1, 32);
            l_i[r] = l_i[r] * corr + ps;
            m_i[r] = mnew;
            o0[r] *= corr;
            o1[r] *= corr;
            s0[r] = p0;
            s1[r] = p1;
        }

        // bounce P through LDS into the A-fragment layout
#pragma unroll
        for (int r = 0; r < 8; ++r) {
            int m = r + ((lane >> 4) << 3);
            Pt[m][lane & 15]        = (f16)s0[r];
            Pt[m][16 + (lane & 15)] = (f16)s1[r];
        }
        __syncthreads();

        v16h pa, vb0, vb1;
#pragma unroll
        for (int e = 0; e < 16; ++e) {
            int kk = ((e >> 3) * 16) + khalf + (e & 7);
            pa[e]  = Pt[mrow][kk];
            vb0[e] = (f16)Vtf[kk][mrow];        // B[k][n] = V[key kk][d n]
            vb1[e] = (f16)Vtf[kk][16 + mrow];
        }
        o0 = __builtin_amdgcn_wmma_f32_16x16x32_f16(false, pa, false, vb0,
                                                    (short)0, o0, false, false);
        o1 = __builtin_amdgcn_wmma_f32_16x16x32_f16(false, pa, false, vb1,
                                                    (short)0, o1, false, false);
        __syncthreads();
    }

#pragma unroll
    for (int r = 0; r < 8; ++r) {
        int m  = r + ((lane >> 4) << 3);
        int gq = q0 + m;
        if (gq >= Lq) continue;
        float inv = 1.0f / (l_i[r] + 1e-20f);
        int d = lane & 15;
        Oh[(size_t)gq * DH + d]      = o0[r] * inv;
        Oh[(size_t)gq * DH + 16 + d] = o1[r] * inv;
    }
}

// ---------------------------------------------------------------------------
// Elementwise / reduction helpers
// ---------------------------------------------------------------------------
__launch_bounds__(256)
__global__ void layernorm_kernel(const float* __restrict__ X,
                                 const float* __restrict__ g,
                                 const float* __restrict__ b,
                                 float* __restrict__ Y, int L)
{
    __shared__ float red[8];
    int row = blockIdx.x;
    int tid = threadIdx.x;
    float v = X[(size_t)row * DMODEL + tid];
    float s = v;
    s += __shfl_xor(s, 16, 32); s += __shfl_xor(s, 8, 32);
    s += __shfl_xor(s, 4, 32);  s += __shfl_xor(s, 2, 32);
    s += __shfl_xor(s, 1, 32);
    if ((tid & 31) == 0) red[tid >> 5] = s;
    __syncthreads();
    float mu = 0.0f;
#pragma unroll
    for (int i = 0; i < 8; ++i) mu += red[i];
    mu *= (1.0f / DMODEL);
    __syncthreads();
    float d  = v - mu;
    float s2 = d * d;
    s2 += __shfl_xor(s2, 16, 32); s2 += __shfl_xor(s2, 8, 32);
    s2 += __shfl_xor(s2, 4, 32);  s2 += __shfl_xor(s2, 2, 32);
    s2 += __shfl_xor(s2, 1, 32);
    if ((tid & 31) == 0) red[tid >> 5] = s2;
    __syncthreads();
    float var = 0.0f;
#pragma unroll
    for (int i = 0; i < 8; ++i) var += red[i];
    var *= (1.0f / DMODEL);
    Y[(size_t)row * DMODEL + tid] = d * rsqrtf(var + 1e-5f) * g[tid] + b[tid];
}

__global__ void add_kernel(const float* A, const float* B, float* Y, int n)
{
    int i = blockIdx.x * blockDim.x + threadIdx.x;
    if (i < n) Y[i] = A[i] + B[i];
}

__global__ void add_bcast_kernel(const float* X, const float* a,
                                 const float* b2, float* Y, int L)
{
    int i = blockIdx.x * blockDim.x + threadIdx.x;
    if (i >= L * DMODEL) return;
    int d = i % DMODEL;
    Y[i] = X[i] + a[d] + b2[d];
}

__global__ void sigmoid_kernel(const float* X, float* Y, int n)
{
    int i = blockIdx.x * blockDim.x + threadIdx.x;
    if (i < n) Y[i] = 1.0f / (1.0f + __expf(-X[i]));
}

__global__ void split_heads_kernel(const float* X, float* Xh, int L)
{
    int i = blockIdx.x * blockDim.x + threadIdx.x;
    if (i >= L * DMODEL) return;
    int l = i / DMODEL, c = i % DMODEL;
    int h = c / DH, d = c % DH;
    Xh[((size_t)h * L + l) * DH + d] = X[i];
}

__global__ void merge_heads_kernel(const float* Xh, float* X, int L)
{
    int i = blockIdx.x * blockDim.x + threadIdx.x;
    if (i >= L * DMODEL) return;
    int l = i / DMODEL, c = i % DMODEL;
    int h = c / DH, d = c % DH;
    X[i] = Xh[((size_t)h * L + l) * DH + d];
}

__global__ void mean_rows_kernel(const float* X, float* out, int L)
{
    int d = threadIdx.x;
    float s = 0.0f;
    for (int l = 0; l < L; ++l) s += X[(size_t)l * DMODEL + d];
    out[d] = s / (float)L;
}

__global__ void softmax16_kernel(float* A, int rows)
{
    int r = blockIdx.x * blockDim.x + threadIdx.x;
    if (r >= rows) return;
    float* p = A + (size_t)r * 16;
    float mx = -1e30f;
#pragma unroll
    for (int i = 0; i < 16; ++i) mx = fmaxf(mx, p[i]);
    float e[16];
    float s = 0.0f;
#pragma unroll
    for (int i = 0; i < 16; ++i) { e[i] = __expf(p[i] - mx); s += e[i]; }
    float inv = 1.0f / s;
#pragma unroll
    for (int i = 0; i < 16; ++i) p[i] = e[i] * inv;
}

// loc = ref.xy + off/NP * ref.wh * 0.5 (valid_ratios == 1)
__global__ void compute_loc_kernel(const float* __restrict__ ref,
                                   const float* __restrict__ off,
                                   float* __restrict__ loc, int Lq)
{
    int i = blockIdx.x * blockDim.x + threadIdx.x;
    int total = Lq * NHEADS * 16;
    if (i >= total) return;
    int q = i / (NHEADS * 16);
    const float* rq = ref + (size_t)q * 4;
    loc[2 * i]     = rq[0] + off[2 * i]     * 0.125f * rq[2];
    loc[2 * i + 1] = rq[1] + off[2 * i + 1] * 0.125f * rq[3];
}

// multi-scale deformable sampling, shapes hardcoded per reference
__global__ void deform_sample_kernel(const float* __restrict__ Vv,
                                     const float* __restrict__ loc,
                                     const float* __restrict__ aw,
                                     float* __restrict__ out, int Lq)
{
    int i = blockIdx.x * blockDim.x + threadIdx.x;
    if (i >= Lq * DMODEL) return;
    int q = i / DMODEL;
    int c = i % DMODEL;
    int h = c / DH;
    const int HW[4] = {28, 14, 7, 4};
    const int ST[4] = {0, 784, 980, 1029};
    float acc = 0.0f;
#pragma unroll
    for (int l = 0; l < 4; ++l) {
        int Hl = HW[l], Wl = HW[l], s = ST[l];
#pragma unroll
        for (int p = 0; p < 4; ++p) {
            size_t li = (((size_t)q * NHEADS + h) * 4 + l) * 4 + p;
            float x = loc[2 * li]     * Wl - 0.5f;
            float y = loc[2 * li + 1] * Hl - 0.5f;
            float w = aw[li];
            int x0 = (int)floorf(x);
            int y0 = (int)floorf(y);
            float fx = x - x0, fy = y - y0;
#pragma unroll
            for (int dy = 0; dy < 2; ++dy)
#pragma unroll
                for (int dx = 0; dx < 2; ++dx) {
                    int xi = x0 + dx, yi = y0 + dy;
                    if (xi < 0 || xi >= Wl || yi < 0 || yi >= Hl) continue;
                    float wgt = (dx ? fx : 1.0f - fx) * (dy ? fy : 1.0f - fy) * w;
                    acc += wgt * Vv[(size_t)(s + yi * Wl + xi) * DMODEL + c];
                }
        }
    }
    out[i] = acc;
}

__global__ void build_enh_kernel(const float* qe, const float* qp,
                                 const float* te, float* enh)
{
    int i = blockIdx.x * blockDim.x + threadIdx.x;
    if (i >= NQT * DMODEL) return;
    int q = i / DMODEL, d = i % DMODEL;
    int t = (q < 256) ? 0 : ((q < 512) ? 1 : 2);
    enh[i] = qe[i] + qp[i] + te[t * DMODEL + d];
}

__global__ void transpose_cls_kernel(const float* in, float* out)
{
    int i = blockIdx.x * blockDim.x + threadIdx.x;
    if (i >= 784 * 10) return;
    int p = i / 10, c = i % 10;
    out[c * 784 + p] = in[i];
}

// ---------------------------------------------------------------------------
// Host orchestration
// ---------------------------------------------------------------------------
namespace {

struct PCur {
    const float* p;
    size_t off;
    const float* take(size_t n) { const float* r = p + off; off += n; return r; }
};

struct WsAlloc {
    char* base;
    size_t off;
    float* falloc(size_t n) {
        float* r = (float*)(base + off);
        off += (n * sizeof(float) + 255) & ~(size_t)255;
        return r;
    }
};

struct MhaBufs { float *qp, *kp, *vp, *qh, *kh, *vh, *oh, *mrg; };

inline void run_linear(hipStream_t s, const float* X, const float* W,
                       const float* b, float* Y, int M, int N, int K, int relu)
{
    dim3 g((M + 63) / 64, (N + 63) / 64);
    linear_wmma_bf16<<<g, 128, 0, s>>>(X, W, b, Y, M, N, K, relu);
}

inline void run_ln(hipStream_t s, const float* X, const float* g,
                   const float* b, float* Y, int L)
{
    layernorm_kernel<<<L, 256, 0, s>>>(X, g, b, Y, L);
}

inline void run_add(hipStream_t s, const float* A, const float* B, float* Y, int n)
{
    add_kernel<<<(n + 255) / 256, 256, 0, s>>>(A, B, Y, n);
}

inline void run_mha(hipStream_t s, MhaBufs& wb,
                    const float* Xq, const float* Xk, const float* Xv,
                    int Lq, int Lk,
                    const float* in_w, const float* in_b,
                    const float* out_w, const float* out_b,
                    float* Y)
{
    const float* wq = in_w;
    const float* wk = in_w + DMODEL * DMODEL;
    const float* wv = in_w + 2 * DMODEL * DMODEL;
    const float* bq = in_b;
    const float* bk = in_b + DMODEL;
    const float* bv = in_b + 2 * DMODEL;
    run_linear(s, Xq, wq, bq, wb.qp, Lq, DMODEL, DMODEL, 0);
    run_linear(s, Xk, wk, bk, wb.kp, Lk, DMODEL, DMODEL, 0);
    run_linear(s, Xv, wv, bv, wb.vp, Lk, DMODEL, DMODEL, 0);
    int g1 = (Lq * DMODEL + 255) / 256;
    int g2 = (Lk * DMODEL + 255) / 256;
    split_heads_kernel<<<g1, 256, 0, s>>>(wb.qp, wb.qh, Lq);
    split_heads_kernel<<<g2, 256, 0, s>>>(wb.kp, wb.kh, Lk);
    split_heads_kernel<<<g2, 256, 0, s>>>(wb.vp, wb.vh, Lk);
    dim3 ag((Lq + 15) / 16, NHEADS);
    flash_attn_wmma<<<ag, 32, 0, s>>>(wb.qh, wb.kh, wb.vh, wb.oh, Lq, Lk,
                                      0.17677669529663687f); // 1/sqrt(32)
    merge_heads_kernel<<<g1, 256, 0, s>>>(wb.oh, wb.mrg, Lq);
    run_linear(s, wb.mrg, out_w, out_b, Y, Lq, DMODEL, DMODEL, 0);
}

} // namespace

extern "C" void kernel_launch(void* const* d_in, const int* in_sizes, int n_in,
                              void* d_out, int out_size, void* d_ws, size_t ws_size,
                              hipStream_t stream)
{
    const float* Vfeat = (const float*)d_in[0];  // [1045][256]
    PCur P{(const float*)d_in[4], 0};
    WsAlloc W{(char*)d_ws, 0};
    float* outp = (float*)d_out;

    const size_t MAXL = 12544;

    // workspace
    float* bufA = W.falloc(MAXL * DMODEL);
    float* bufB = W.falloc(MAXL * DMODEL);
    float* bufC = W.falloc(MAXL * DMODEL);
    float* bufD = W.falloc(MAXL * DMODEL);
    MhaBufs wb;
    wb.qp = W.falloc(MAXL * DMODEL); wb.kp = W.falloc(MAXL * DMODEL);
    wb.vp = W.falloc(MAXL * DMODEL); wb.qh = W.falloc(MAXL * DMODEL);
    wb.kh = W.falloc(MAXL * DMODEL); wb.vh = W.falloc(MAXL * DMODEL);
    wb.oh = W.falloc(MAXL * DMODEL); wb.mrg = W.falloc(MAXL * DMODEL);
    float* ffbuf  = W.falloc(MAXL * 512);
    float* curbuf = W.falloc(MAXL * DMODEL);
    float* enh    = W.falloc((size_t)NQT * DMODEL);
    float* refbuf = W.falloc((size_t)NQT * 4);
    float* offb   = W.falloc((size_t)NQT * DMODEL);
    float* locb   = W.falloc((size_t)NQT * DMODEL);
    float* awbuf  = W.falloc((size_t)NQT * 128);
    float* vval   = W.falloc((size_t)LVTOT * DMODEL);
    float* dsamp  = W.falloc((size_t)NQT * DMODEL);
    float* seg2b  = W.falloc(256 * DMODEL);
    float* det2b  = W.falloc(256 * DMODEL);
    float* mean1  = W.falloc(DMODEL);
    float* mean2  = W.falloc(DMODEL);
    float* s2pb   = W.falloc(DMODEL);
    float* d2pb   = W.falloc(DMODEL);
    float* pteb   = W.falloc(3 * DMODEL);
    float* membuf = W.falloc((size_t)(256 + LVTOT) * DMODEL);
    float* tgtbuf = W.falloc(100 * DMODEL);
    float* logits = W.falloc(784 * 10);

    // ---------------- parameters (deterministic cursor) ----------------
    const float* query_embed = P.take((size_t)NQT * DMODEL);
    const float* query_pos   = P.take((size_t)NQT * DMODEL);
    const float* task_embed  = P.take(3 * DMODEL);
    const float* refW = P.take(4 * DMODEL);
    const float* refB = P.take(4);

    struct DecL {
        const float *saw, *sab, *saow, *saob;
        const float *n1g, *n1b, *n2g, *n2b, *n3g, *n3b;
        const float *sow, *sob, *aww, *awb, *vpw, *vpb, *opw, *opb;
        const float *f1w, *f1b, *f2w, *f2b;
    } dl[6];
    for (int i = 0; i < 6; ++i) {
        dl[i].saw = P.take(3 * DMODEL * DMODEL); dl[i].sab = P.take(3 * DMODEL);
        dl[i].saow = P.take(DMODEL * DMODEL);    dl[i].saob = P.take(DMODEL);
        dl[i].n1g = P.take(DMODEL); dl[i].n1b = P.take(DMODEL);
        dl[i].n2g = P.take(DMODEL); dl[i].n2b = P.take(DMODEL);
        dl[i].n3g = P.take(DMODEL); dl[i].n3b = P.take(DMODEL);
        dl[i].sow = P.take(DMODEL * DMODEL);  dl[i].sob = P.take(DMODEL);
        dl[i].aww = P.take(128 * DMODEL);     dl[i].awb = P.take(128);
        dl[i].vpw = P.take(DMODEL * DMODEL);  dl[i].vpb = P.take(DMODEL);
        dl[i].opw = P.take(DMODEL * DMODEL);  dl[i].opb = P.take(DMODEL);
        dl[i].f1w = P.take(1024 * DMODEL);    dl[i].f1b = P.take(1024);
        dl[i].f2w = P.take(DMODEL * 1024);    dl[i].f2b = P.take(DMODEL);
    }
    const float* ct_iw = P.take(3 * DMODEL * DMODEL);
    const float* ct_ib = P.take(3 * DMODEL);
    const float* ct_ow = P.take(DMODEL * DMODEL);
    const float* ct_ob = P.take(DMODEL);
    const float* ctng = P.take(DMODEL);
    const float* ctnb = P.take(DMODEL);
    const float* seg1w = P.take(DMODEL * DMODEL); const float* seg1b = P.take(DMODEL);
    const float* seg2w = P.take(DMODEL * DMODEL); const float* seg2bi = P.take(DMODEL);
    const float* dc1w = P.take(DMODEL * DMODEL); const float* dc1b = P.take(DMODEL);
    const float* dc2w = P.take(11 * DMODEL);     const float* dc2b = P.take(11);
    const float* db1w = P.take(DMODEL * DMODEL); const float* db1b = P.take(DMODEL);
    const float* db2w = P.take(4 * DMODEL);      const float* db2b = P.take(4);
    const float* pc1w = P.take(DMODEL * DMODEL); const float* pc1b = P.take(DMODEL);
    const float* pc2w = P.take(14 * DMODEL);     const float* pc2b = P.take(14);
    const float* s2pw = P.take(DMODEL * DMODEL); const float* s2pbias = P.take(DMODEL);
    const float* d2pw = P.take(DMODEL * DMODEL); const float* d2pbias = P.take(DMODEL);

    // pixel decoder params
    const float* pix_seg_q   = P.take(100 * DMODEL);
    const float* pix_seg_pos = P.take(100 * DMODEL);
    struct TDecL {
        const float *sa_iw, *sa_ib, *sa_ow, *sa_ob;
        const float *ca_iw, *ca_ib, *ca_ow, *ca_ob;
        const float *n1g, *n1b, *n2g, *n2b, *n3g, *n3b;
        const float *f1w, *f1b, *f2w, *f2b;
    } pdl[2];
    for (int i = 0; i < 2; ++i) {
        pdl[i].sa_iw = P.take(3 * DMODEL * DMODEL); pdl[i].sa_ib = P.take(3 * DMODEL);
        pdl[i].sa_ow = P.take(DMODEL * DMODEL);     pdl[i].sa_ob = P.take(DMODEL);
        pdl[i].ca_iw = P.take(3 * DMODEL * DMODEL); pdl[i].ca_ib = P.take(3 * DMODEL);
        pdl[i].ca_ow = P.take(DMODEL * DMODEL);     pdl[i].ca_ob = P.take(DMODEL);
        pdl[i].n1g = P.take(DMODEL); pdl[i].n1b = P.take(DMODEL);
        pdl[i].n2g = P.take(DMODEL); pdl[i].n2b = P.take(DMODEL);
        pdl[i].n3g = P.take(DMODEL); pdl[i].n3b = P.take(DMODEL);
        pdl[i].f1w = P.take(512 * DMODEL); pdl[i].f1b = P.take(512);
        pdl[i].f2w = P.take(DMODEL * 512); pdl[i].f2b = P.take(DMODEL);
    }
    const float* fuse1w = P.take(DMODEL * DMODEL); const float* fuse1b = P.take(DMODEL);
    const float* fuse2w = P.take(DMODEL * DMODEL); const float* fuse2b = P.take(DMODEL);
    struct StgL {
        const float *ca_iw, *ca_ib, *ca_ow, *ca_ob;
        const float *sa_iw, *sa_ib, *sa_ow, *sa_ob;
        const float *n1g, *n1b, *n2g, *n2b, *n3g, *n3b;
        const float *f1w, *f1b, *f2w, *f2b;
    } st[3];
    for (int i = 0; i < 3; ++i) {
        st[i].ca_iw = P.take(3 * DMODEL * DMODEL); st[i].ca_ib = P.take(3 * DMODEL);
        st[i].ca_ow = P.take(DMODEL * DMODEL);     st[i].ca_ob = P.take(DMODEL);
        st[i].sa_iw = P.take(3 * DMODEL * DMODEL); st[i].sa_ib = P.take(3 * DMODEL);
        st[i].sa_ow = P.take(DMODEL * DMODEL);     st[i].sa_ob = P.take(DMODEL);
        st[i].n1g = P.take(DMODEL); st[i].n1b = P.take(DMODEL);
        st[i].n2g = P.take(DMODEL); st[i].n2b = P.take(DMODEL);
        st[i].n3g = P.take(DMODEL); st[i].n3b = P.take(DMODEL);
        st[i].f1w = P.take(512 * DMODEL); st[i].f1b = P.take(512);
        st[i].f2w = P.take(DMODEL * 512); st[i].f2b = P.take(DMODEL);
    }
    const int STAGE_L[3] = {3136, 12544, 784};
    const float* pq0 = P.take((size_t)3136 * DMODEL);
    const float* pq1 = P.take((size_t)12544 * DMODEL);
    const float* pq2 = P.take((size_t)784 * DMODEL);
    const float* pqs[3] = {pq0, pq1, pq2};
    const float* cls1w = P.take(128 * DMODEL); const float* cls1b = P.take(128);
    const float* cls2w = P.take(10 * 128);     const float* cls2b = P.take(10);

    const int nQ = NQT * DMODEL;
    const int gQ = (nQ + 255) / 256;

    // ---------------- forward ----------------
    build_enh_kernel<<<gQ, 256, 0, stream>>>(query_embed, query_pos, task_embed, enh);
    run_linear(stream, enh, refW, refB, refbuf, NQT, 4, DMODEL, 0);
    sigmoid_kernel<<<(NQT * 4 + 255) / 256, 256, 0, stream>>>(refbuf, refbuf, NQT * 4);
    (void)hipMemcpyAsync(bufA, enh, (size_t)nQ * 4, hipMemcpyDeviceToDevice, stream);

    // 6 deformable decoder layers
    for (int i = 0; i < 6; ++i) {
        run_add(stream, bufA, query_pos, bufB, nQ);
        run_mha(stream, wb, bufB, bufB, bufA, NQT, NQT,
                dl[i].saw, dl[i].sab, dl[i].saow, dl[i].saob, bufC);
        run_add(stream, bufA, bufC, bufD, nQ);
        run_ln(stream, bufD, dl[i].n2g, dl[i].n2b, bufA, NQT);

        run_add(stream, bufA, query_pos, bufB, nQ);
        run_linear(stream, Vfeat, dl[i].vpw, dl[i].vpb, vval, LVTOT, DMODEL, DMODEL, 0);
        run_linear(stream, bufB, dl[i].sow, dl[i].sob, offb, NQT, DMODEL, DMODEL, 0);
        run_linear(stream, bufB, dl[i].aww, dl[i].awb, awbuf, NQT, 128, DMODEL, 0);
        softmax16_kernel<<<(NQT * 8 + 255) / 256, 256, 0, stream>>>(awbuf, NQT * 8);
        compute_loc_kernel<<<(NQT * 128 + 255) / 256, 256, 0, stream>>>(refbuf, offb, locb, NQT);
        deform_sample_kernel<<<gQ, 256, 0, stream>>>(vval, locb, awbuf, dsamp, NQT);
        run_linear(stream, dsamp, dl[i].opw, dl[i].opb, bufC, NQT, DMODEL, DMODEL, 0);
        run_add(stream, bufA, bufC, bufD, nQ);
        run_ln(stream, bufD, dl[i].n1g, dl[i].n1b, bufA, NQT);

        run_linear(stream, bufA, dl[i].f1w, dl[i].f1b, ffbuf, NQT, 1024, DMODEL, 1);
        run_linear(stream, ffbuf, dl[i].f2w, dl[i].f2b, bufC, NQT, DMODEL, 1024, 0);
        run_add(stream, bufA, bufC, bufD, nQ);
        run_ln(stream, bufD, dl[i].n3g, dl[i].n3b, bufA, NQT);
    }

    // cross-task attention
    const float* seg_f = bufA;
    const float* det_f = bufA + 256 * DMODEL;
    const float* pt_f  = bufA + 512 * DMODEL;
    run_mha(stream, wb, seg_f, det_f, det_f, 256, 256, ct_iw, ct_ib, ct_ow, ct_ob, bufC);
    run_add(stream, seg_f, bufC, bufD, 256 * DMODEL);
    run_ln(stream, bufD, ctng, ctnb, seg2b, 256);
    run_mha(stream, wb, det_f, seg2b, seg2b, 256, 256, ct_iw, ct_ib, ct_ow, ct_ob, bufC);
    run_add(stream, det_f, bufC, bufD, 256 * DMODEL);
    run_ln(stream, bufD, ctng, ctnb, det2b, 256);

    mean_rows_kernel<<<1, 256, 0, stream>>>(seg2b, mean1, 256);
    run_linear(stream, mean1, s2pw, s2pbias, s2pb, 1, DMODEL, DMODEL, 0);
    mean_rows_kernel<<<1, 256, 0, stream>>>(det2b, mean2, 256);
    run_linear(stream, mean2, d2pw, d2pbias, d2pb, 1, DMODEL, DMODEL, 0);
    add_bcast_kernel<<<(3 * DMODEL + 255) / 256, 256, 0, stream>>>(pt_f, s2pb, d2pb, pteb, 3);

    // seg_qf -> d_out[0 .. 65536)
    run_linear(stream, seg2b, seg1w, seg1b, bufB, 256, DMODEL, DMODEL, 1);
    run_linear(stream, bufB, seg2w, seg2bi, outp, 256, DMODEL, DMODEL, 0);

    // det heads -> d_out
    float* o_cls = outp + 65536 + 7840;
    float* o_box = o_cls + 2816;
    float* o_pt  = o_box + 1024;
    run_linear(stream, det2b, dc1w, dc1b, bufB, 256, DMODEL, DMODEL, 1);
    run_linear(stream, bufB, dc2w, dc2b, o_cls, 256, 11, DMODEL, 0);
    run_linear(stream, det2b, db1w, db1b, bufB, 256, DMODEL, DMODEL, 1);
    run_linear(stream, bufB, db2w, db2b, bufC, 256, 4, DMODEL, 0);
    sigmoid_kernel<<<(1024 + 255) / 256, 256, 0, stream>>>(bufC, o_box, 1024);
    run_linear(stream, pteb, pc1w, pc1b, bufB, 3, DMODEL, DMODEL, 1);
    run_linear(stream, bufB, pc2w, pc2b, o_pt, 3, 14, DMODEL, 0);

    // ---------------- pixel decoder ----------------
    const int MEML = 256 + LVTOT;
    (void)hipMemcpyAsync(membuf, outp, (size_t)256 * DMODEL * 4,
                         hipMemcpyDeviceToDevice, stream);
    (void)hipMemcpyAsync(membuf + 256 * DMODEL, Vfeat, (size_t)LVTOT * DMODEL * 4,
                         hipMemcpyDeviceToDevice, stream);
    run_add(stream, pix_seg_q, pix_seg_pos, tgtbuf, 100 * DMODEL);

    for (int i = 0; i < 2; ++i) {
        run_mha(stream, wb, tgtbuf, tgtbuf, tgtbuf, 100, 100,
                pdl[i].sa_iw, pdl[i].sa_ib, pdl[i].sa_ow, pdl[i].sa_ob, bufC);
        run_add(stream, tgtbuf, bufC, bufD, 100 * DMODEL);
        run_ln(stream, bufD, pdl[i].n1g, pdl[i].n1b, tgtbuf, 100);
        run_mha(stream, wb, tgtbuf, membuf, membuf, 100, MEML,
                pdl[i].ca_iw, pdl[i].ca_ib, pdl[i].ca_ow, pdl[i].ca_ob, bufC);
        run_add(stream, tgtbuf, bufC, bufD, 100 * DMODEL);
        run_ln(stream, bufD, pdl[i].n2g, pdl[i].n2b, tgtbuf, 100);
        run_linear(stream, tgtbuf, pdl[i].f1w, pdl[i].f1b, ffbuf, 100, 512, DMODEL, 1);
        run_linear(stream, ffbuf, pdl[i].f2w, pdl[i].f2b, bufC, 100, DMODEL, 512, 0);
        run_add(stream, tgtbuf, bufC, bufD, 100 * DMODEL);
        run_ln(stream, bufD, pdl[i].n3g, pdl[i].n3b, tgtbuf, 100);
    }
    run_linear(stream, tgtbuf, fuse1w, fuse1b, bufB, 100, DMODEL, DMODEL, 1);
    run_linear(stream, bufB, fuse2w, fuse2b, curbuf, 100, DMODEL, DMODEL, 0);
    int curL = 100;

    for (int i = 0; i < 3; ++i) {
        int Ls = STAGE_L[i];
        int nS = Ls * DMODEL;
        run_mha(stream, wb, pqs[i], curbuf, curbuf, Ls, curL,
                st[i].ca_iw, st[i].ca_ib, st[i].ca_ow, st[i].ca_ob, bufC);
        run_add(stream, pqs[i], bufC, bufD, nS);
        run_ln(stream, bufD, st[i].n1g, st[i].n1b, bufA, Ls);
        run_mha(stream, wb, bufA, bufA, bufA, Ls, Ls,
                st[i].sa_iw, st[i].sa_ib, st[i].sa_ow, st[i].sa_ob, bufC);
        run_add(stream, bufA, bufC, bufD, nS);
        run_ln(stream, bufD, st[i].n2g, st[i].n2b, bufB, Ls);
        run_linear(stream, bufB, st[i].f1w, st[i].f1b, ffbuf, Ls, 512, DMODEL, 1);
        run_linear(stream, ffbuf, st[i].f2w, st[i].f2b, bufC, Ls, DMODEL, 512, 0);
        run_add(stream, bufB, bufC, bufD, nS);
        run_ln(stream, bufD, st[i].n3g, st[i].n3b, curbuf, Ls);
        curL = Ls;
    }

    // seg logits -> d_out[65536 .. 65536+7840)
    run_linear(stream, curbuf, cls1w, cls1b, bufB, 784, 128, DMODEL, 1);
    run_linear(stream, bufB, cls2w, cls2b, logits, 784, 10, 128, 0);
    transpose_cls_kernel<<<(7840 + 255) / 256, 256, 0, stream>>>(logits, outp + 65536);

    (void)in_sizes; (void)n_in; (void)out_size; (void)ws_size;
}
